// SparseGaussianSelfAttention_81346680586522
// MI455X (gfx1250) — compile-verified, matrix operand order checked by reading, not run on hardware
//
#include <hip/hip_runtime.h>

// ---------------------------------------------------------------------------
// SparseGaussianSelfAttention for MI455X (gfx1250, wave32)
//   N=50000 nodes, D=256, H=8 heads, DH=32, K=17 neighbors
//   GEMMs (26 GFLOP) -> v_wmma_f32_16x16x32_bf16, weights pre-packed to
//   fragment-ordered bf16 so the inner loop is pure {b128 load, wmma}.
//   Gather/attention (1.7GB reads) -> bf16 QKV (76.8MB) stays in 192MB L2.
//   proj_ln stages its bf16 stripe via GLOBAL_LOAD_ASYNC_TO_LDS_B128
//   (ASYNCcnt path) and fuses bias+residual+LayerNorm in LDS.
// ---------------------------------------------------------------------------

#define NN    50000
#define DD    256
#define HH    8
#define DHH   32
#define KK    17
#define STRIPES (NN / 16)   // 3125, exact

typedef __attribute__((ext_vector_type(16))) __bf16         v16bf;
typedef __attribute__((ext_vector_type(8)))  float          v8f;
typedef __attribute__((ext_vector_type(8)))  unsigned short v8u;

union Frag16 {
    v16bf bf;
    v8u   h[2];
    unsigned short e[16];
};

// fp32 -> bf16 bits, round-to-nearest-even (bit-level; only used OUTSIDE the
// GEMM inner loops: weight prep, feature staging, output stores)
__device__ __forceinline__ unsigned short f2u(float f) {
    union { float f; unsigned u; } a; a.f = f;
    unsigned r = (a.u + 0x7fffu + ((a.u >> 16) & 1u)) >> 16;
    return (unsigned short)r;
}
__device__ __forceinline__ float u2f(unsigned short s) {
    union { unsigned u; float f; } o; o.u = ((unsigned)s) << 16;
    return o.f;
}

__device__ __forceinline__ v8f wmma_bf16(const Frag16& a, const Frag16& b,
                                         v8f c) {
    return __builtin_amdgcn_wmma_f32_16x16x32_bf16(
        /*neg_a=*/false, a.bf, /*neg_b=*/false, b.bf,
        /*c_mod=*/(short)0, c, /*reuse_a=*/false, /*reuse_b=*/false);
}

// ---------------------------------------------------------------------------
// Kernel 0: pack {Wq,Wk,Wv,Wo} (fp32 row-major, out = A @ W^T) into
// fragment-ordered bf16: pack[(((m*16+ct)*8+kk)*32+lane)*16 + e].
// ---------------------------------------------------------------------------
__global__ __launch_bounds__(256) void pack_weights(
    const float* __restrict__ Wq, const float* __restrict__ Wk,
    const float* __restrict__ Wv, const float* __restrict__ Wo,
    unsigned short* __restrict__ pack) {
    const int gid = blockIdx.x * 256 + threadIdx.x;   // 0..16383
    const int m   = gid >> 12;                        // matrix
    const int rem = gid & 4095;
    const int l   = rem & 31;                         // lane
    const int kk  = (rem >> 5) & 7;                   // k-step
    const int ct  = rem >> 8;                         // column tile

    const float* W = (m == 0) ? Wq : (m == 1) ? Wk : (m == 2) ? Wv : Wo;
    const int n  = ct * 16 + (l & 15);
    const int hi = (l >> 4) * 8;
    const float* p = W + n * DD + kk * 32 + hi;
    unsigned short* dst = pack + (size_t)gid * 16;
#pragma unroll
    for (int j = 0; j < 8; ++j) {
        dst[j]     = f2u(p[j]);
        dst[8 + j] = f2u(p[16 + j]);
    }
}

// ---------------------------------------------------------------------------
// Kernel 1: Q/K/V projections.  Block per 16-row stripe, 8 waves.
// A fragments hoisted to registers; job selection on SALU via readfirstlane;
// inner loop = 2x global b128 (packed B) + wmma.
// ---------------------------------------------------------------------------
__global__ __launch_bounds__(256) void qkv_gemm(
    const float* __restrict__ feat,
    const float* __restrict__ bq, const float* __restrict__ bk,
    const float* __restrict__ bv,
    const unsigned short* __restrict__ wpack,
    unsigned short* __restrict__ Qb, unsigned short* __restrict__ Kb,
    unsigned short* __restrict__ Vb) {
    __shared__ unsigned short Abuf[16 * DD];   // 8 KB stripe in bf16

    const int stripe = blockIdx.x;
    const int t = threadIdx.x;
    const size_t rowbase = (size_t)stripe * 16 * DD;

#pragma unroll
    for (int i = 0; i < 16; ++i) {
        int idx = t + i * 256;
        Abuf[idx] = f2u(feat[rowbase + idx]);
    }
    __syncthreads();

    const int w   = __builtin_amdgcn_readfirstlane(t >> 5);  // wave-uniform
    const int l   = t & 31;
    const int m16 = l & 15;
    const int hi  = (l >> 4) * 8;

    // hoist 8 A fragments into registers (reused by all 6 jobs)
    Frag16 af[8];
#pragma unroll
    for (int kk = 0; kk < 8; ++kk) {
        const unsigned short* ap = &Abuf[m16 * DD + kk * 32 + hi];
        af[kk].h[0] = *(const v8u*)(ap);
        af[kk].h[1] = *(const v8u*)(ap + 16);
    }

#pragma unroll
    for (int jj = 0; jj < 6; ++jj) {
        const int job = w * 6 + jj;     // scalar
        const int mat = job >> 4;       // 0=Q 1=K 2=V   (scalar)
        const int ct  = job & 15;       // column tile   (scalar)
        const float* bs = (mat == 0) ? bq : (mat == 1) ? bk : bv;
        unsigned short* Ob = (mat == 0) ? Qb : (mat == 1) ? Kb : Vb;
        const unsigned short* wp =
            wpack + ((size_t)((mat * 16 + ct) * 8) * 32 + l) * 16;

        v8f acc = {};
#pragma unroll
        for (int kk = 0; kk < 8; ++kk) {
            Frag16 b;
            const unsigned short* bp = wp + (size_t)kk * 32 * 16;
            b.h[0] = *(const v8u*)(bp);
            b.h[1] = *(const v8u*)(bp + 8);
            acc = wmma_bf16(af[kk], b, acc);
        }

        const int n = ct * 16 + m16;
        const float bias = bs[n];
#pragma unroll
        for (int r = 0; r < 8; ++r) {
            const int row = stripe * 16 + hi + r;   // M = r + 8*(lane>=16)
            Ob[(size_t)row * DD + n] = f2u(acc[r] + bias);
        }
    }
}

// ---------------------------------------------------------------------------
// Kernel 2: sparse neighbor attention.  Block = node, wave = head,
// lane = head dim.  Q/K/V bf16 live in L2 (76.8MB < 192MB).
// ---------------------------------------------------------------------------
__global__ __launch_bounds__(256) void attn_kernel(
    const unsigned short* __restrict__ Qb,
    const unsigned short* __restrict__ Kb,
    const unsigned short* __restrict__ Vb,
    const int* __restrict__ knn,
    const unsigned char* __restrict__ valid,
    unsigned short* __restrict__ Ab) {
    const int n = blockIdx.x;
    const int h = threadIdx.x >> 5;
    const int l = threadIdx.x & 31;

    const size_t qoff = (size_t)n * DD + h * DHH + l;
    const float q = u2f(Qb[qoff]);

    int   idxs[KK];
    float lg[KK];
    const float scale = 0.17677669529663687f;   // 1/sqrt(32)

#pragma unroll
    for (int k = 0; k < KK; ++k) {
        const int idx = knn[n * KK + k];        // uniform -> scalar load
        idxs[k] = idx;
        const float kv = u2f(Kb[(size_t)idx * DD + h * DHH + l]);
        float p = q * kv;
        p += __shfl_xor(p, 16, 32);
        p += __shfl_xor(p, 8, 32);
        p += __shfl_xor(p, 4, 32);
        p += __shfl_xor(p, 2, 32);
        p += __shfl_xor(p, 1, 32);
        lg[k] = valid[n * KK + k] ? p * scale : -1e9f;
    }

    float mx = lg[0];
#pragma unroll
    for (int k = 1; k < KK; ++k) mx = fmaxf(mx, lg[k]);
    float s = 0.0f;
#pragma unroll
    for (int k = 0; k < KK; ++k) { lg[k] = __expf(lg[k] - mx); s += lg[k]; }
    const float inv = 1.0f / s;

    float acc = 0.0f;
#pragma unroll
    for (int k = 0; k < KK; ++k) {
        const float vv = u2f(Vb[(size_t)idxs[k] * DD + h * DHH + l]);
        acc += (lg[k] * inv) * vv;
    }
    Ab[qoff] = f2u(acc);
}

// ---------------------------------------------------------------------------
// Kernel 3: output projection (WMMA) + bias + residual + LayerNorm, fused.
// Attn stripe staged global->LDS via async-to-LDS (ASYNCcnt), no VGPR trip.
// ---------------------------------------------------------------------------
__global__ __launch_bounds__(256) void proj_ln_kernel(
    const unsigned short* __restrict__ Ab,
    const unsigned short* __restrict__ wpack,   // Wo section (m=3)
    const float* __restrict__ bo,
    const float* __restrict__ feat,
    const float* __restrict__ gamma, const float* __restrict__ beta,
    float* __restrict__ out) {
    __shared__ unsigned short Sa[16 * DD];   // 8 KB attn stripe bf16
    __shared__ float X[16 * DD];             // 16 KB pre-LN fp32

    const int stripe = blockIdx.x;
    const int t = threadIdx.x;
    const size_t rowbase = (size_t)stripe * 16 * DD;

    {   // async copy 8KB stripe: global -> LDS, 2 x b128 per thread.
        // INST_OFFSET applies to both LDS and global addresses (ISA 08 §4.4).
        unsigned long long g =
            (unsigned long long)(const void*)(Ab + rowbase)
            + (unsigned long long)t * 16ull;
        unsigned lds = (unsigned)(unsigned long long)(const void*)&Sa[0]
                       + (unsigned)t * 16u;
        asm volatile("global_load_async_to_lds_b128 %0, %1, off"
                     :: "v"(lds), "v"(g) : "memory");
        asm volatile("global_load_async_to_lds_b128 %0, %1, off offset:4096"
                     :: "v"(lds), "v"(g) : "memory");
        asm volatile("s_wait_asynccnt 0x0" ::: "memory");
    }
    __syncthreads();

    const int w   = __builtin_amdgcn_readfirstlane(t >> 5);  // wave-uniform
    const int l   = t & 31;
    const int m16 = l & 15;
    const int hi  = (l >> 4) * 8;

    Frag16 af[8];
#pragma unroll
    for (int kk = 0; kk < 8; ++kk) {
        const unsigned short* ap = &Sa[m16 * DD + kk * 32 + hi];
        af[kk].h[0] = *(const v8u*)(ap);
        af[kk].h[1] = *(const v8u*)(ap + 16);
    }

#pragma unroll
    for (int j = 0; j < 2; ++j) {
        const int ct = w + j * 8;       // scalar
        const unsigned short* wp =
            wpack + ((size_t)((3 * 16 + ct) * 8) * 32 + l) * 16;

        v8f acc = {};
#pragma unroll
        for (int kk = 0; kk < 8; ++kk) {
            Frag16 b;
            const unsigned short* bp = wp + (size_t)kk * 32 * 16;
            b.h[0] = *(const v8u*)(bp);
            b.h[1] = *(const v8u*)(bp + 8);
            acc = wmma_bf16(af[kk], b, acc);
        }
        const int n = ct * 16 + m16;
        const float bias = bo[n];
#pragma unroll
        for (int r = 0; r < 8; ++r) {
            const int row = hi + r;
            X[row * DD + n] = acc[r] + bias + feat[rowbase + row * DD + n];
        }
    }
    __syncthreads();

#pragma unroll
    for (int rr = 0; rr < 2; ++rr) {
        const int row = w * 2 + rr;
        float s = 0.0f, s2 = 0.0f;
#pragma unroll
        for (int j = 0; j < 8; ++j) {
            const float x = X[row * DD + l + j * 32];
            s += x; s2 += x * x;
        }
        for (int off = 16; off; off >>= 1) {
            s  += __shfl_xor(s,  off, 32);
            s2 += __shfl_xor(s2, off, 32);
        }
        const float mu   = s * (1.0f / 256.0f);
        const float var  = s2 * (1.0f / 256.0f) - mu * mu;
        const float rstd = rsqrtf(var + 1e-5f);
#pragma unroll
        for (int j = 0; j < 8; ++j) {
            const int c = l + j * 32;
            const float x = X[row * DD + c];
            out[rowbase + row * DD + c] = (x - mu) * rstd * gamma[c] + beta[c];
        }
    }
}

// ---------------------------------------------------------------------------
extern "C" void kernel_launch(void* const* d_in, const int* in_sizes, int n_in,
                              void* d_out, int out_size, void* d_ws,
                              size_t ws_size, hipStream_t stream) {
    (void)in_sizes; (void)n_in; (void)out_size; (void)ws_size;

    const float* feat  = (const float*)d_in[0];
    const float* Wq    = (const float*)d_in[1];
    const float* bq    = (const float*)d_in[2];
    const float* Wk    = (const float*)d_in[3];
    const float* bk    = (const float*)d_in[4];
    const float* Wv    = (const float*)d_in[5];
    const float* bv    = (const float*)d_in[6];
    const float* Wo    = (const float*)d_in[7];
    const float* bo    = (const float*)d_in[8];
    const float* gamma = (const float*)d_in[9];
    const float* beta  = (const float*)d_in[10];
    const int*   knn   = (const int*)d_in[11];
    const unsigned char* valid = (const unsigned char*)d_in[12];
    float* out = (float*)d_out;

    const size_t ND = (size_t)NN * DD;
    unsigned short* Qb    = (unsigned short*)d_ws;   // 25.6 MB each
    unsigned short* Kb    = Qb + ND;
    unsigned short* Vb    = Kb + ND;
    unsigned short* Ab    = Vb + ND;
    unsigned short* wpack = Ab + ND;                 // +512 KB packed weights

    pack_weights<<<64, 256, 0, stream>>>(Wq, Wk, Wv, Wo, wpack);
    qkv_gemm<<<STRIPES, 256, 0, stream>>>(feat, bq, bk, bv, wpack,
                                          Qb, Kb, Vb);
    attn_kernel<<<NN, 256, 0, stream>>>(Qb, Kb, Vb, knn, valid, Ab);
    proj_ln_kernel<<<STRIPES, 256, 0, stream>>>(Ab, wpack, bo, feat,
                                                gamma, beta, out);
}
